// AutoEncoderTopK_29695403885147
// MI455X (gfx1250) — matrix operand order repaired and checked by reading, not compiled, not run on hardware
//
#include <hip/hip_runtime.h>

typedef __bf16 bf16;
typedef __attribute__((ext_vector_type(16))) __bf16 v16bf;
typedef __attribute__((ext_vector_type(8)))  float  v8f;

#define D_IN   2048
#define D_SAE  16384
#define KTOP   100
#define NBATCH 8192

// ---------------------------------------------------------------------------
// CDNA5 async global->LDS copy (ASYNCcnt-tracked DMA, bypasses VGPRs)
//   lds_off : workgroup-relative LDS byte offset (low 32 bits of generic ptr)
//   gptr    : 64-bit global VA
// ---------------------------------------------------------------------------
__device__ __forceinline__ void async_copy16(unsigned lds_off, const void* gptr) {
    asm volatile("global_load_async_to_lds_b128 %0, %1, off"
                 :: "v"(lds_off), "v"(gptr)
                 : "memory");
}
__device__ __forceinline__ void wait_async0() {
#if __has_builtin(__builtin_amdgcn_s_wait_asynccnt)
    __builtin_amdgcn_s_wait_asynccnt(0);
#else
    asm volatile("s_wait_asynccnt 0x0" ::: "memory");
#endif
}
__device__ __forceinline__ unsigned lds_off32(const void* p) {
    // generic pointers to LDS carry the workgroup-relative offset in addr[31:0]
    return (unsigned)(uintptr_t)p;
}

// ---------------------------------------------------------------------------
// Kernel 1: xb = bf16(x - b_dec), streaming, float4 in / 4x bf16 out
// ---------------------------------------------------------------------------
__global__ __launch_bounds__(256) void k_convert_x(const float* __restrict__ x,
                                                   const float* __restrict__ bdec,
                                                   bf16* __restrict__ xb) {
    const int i4   = blockIdx.x * 256 + threadIdx.x;
    const int base = i4 * 4;
    const float4 v  = *(const float4*)(x + base);
    const int col   = base & (D_IN - 1);
    const float4 bd = *(const float4*)(bdec + col);
    __attribute__((aligned(8))) bf16 o[4];
    o[0] = (bf16)(v.x - bd.x);
    o[1] = (bf16)(v.y - bd.y);
    o[2] = (bf16)(v.z - bd.z);
    o[3] = (bf16)(v.w - bd.w);
    *(uint2*)(xb + base) = *(const uint2*)o;
}

// ---------------------------------------------------------------------------
// Kernel 2: W_encT[n][k] = bf16(W_enc[k][n])  (LDS-tiled 64x64 transpose)
// ---------------------------------------------------------------------------
__global__ __launch_bounds__(256) void k_convert_wT(const float* __restrict__ W,
                                                    bf16* __restrict__ WT) {
    __shared__ bf16 tile[64][66];           // padded to dodge bank conflicts
    const int t  = threadIdx.x;
    const int n0 = blockIdx.x * 64;
    const int k0 = blockIdx.y * 64;
#pragma unroll
    for (int i = 0; i < 16; ++i) {          // read coalesced along n
        const int id = t + i * 256;
        const int kl = id >> 6, nl = id & 63;
        tile[kl][nl] = (bf16)W[(size_t)(k0 + kl) * D_SAE + (n0 + nl)];
    }
    __syncthreads();
#pragma unroll
    for (int i = 0; i < 16; ++i) {          // write coalesced along k
        const int id = t + i * 256;
        const int nl = id >> 6, kl = id & 63;
        WT[(size_t)(n0 + nl) * D_IN + (k0 + kl)] = tile[kl][nl];
    }
}

// ---------------------------------------------------------------------------
// Kernel 3: encode GEMM (bf16 WMMA, fp32 acc) + b_enc + ReLU -> pre_acts
//   A = xb   [NBATCH][D_IN]  bf16 row-major
//   B = WencT[D_SAE ][D_IN]  bf16 row-major  (i.e. W_enc transposed)
//   Block tile 128(M) x 128(N) x 32(K); 8 waves as 2(M) x 4(N);
//   wave tile 64x32 = 4x2 subtiles of 16x16; double-buffered LDS fed by
//   global_load_async_to_lds_b128 (one barrier per K-step).
// ---------------------------------------------------------------------------
__global__ __launch_bounds__(256) void k_encode_gemm(const bf16* __restrict__ A,
                                                     const bf16* __restrict__ Bm,
                                                     const float* __restrict__ benc,
                                                     float* __restrict__ pre) {
    __shared__ bf16 As[2][128][32];         // 8 KB per buffer
    __shared__ bf16 Bs[2][128][32];

    const int t    = threadIdx.x;
    const int lane = t & 31;
    const int wid  = t >> 5;
    const int wm   = wid >> 2;      // 0..1  -> M offset wm*64
    const int wn   = wid & 3;       // 0..3  -> N offset wn*32
    const int m0   = blockIdx.y * 128;
    const int n0   = blockIdx.x * 128;

    // async staging: 512 chunks of 16B per tile, 2 per thread per matrix
    const int lr = t >> 2;          // 0..63  (row, +64 for second chunk)
    const int ls = (t & 3) * 8;     // 0,8,16,24 (bf16 col within 32)
    const bf16* Ab = A  + (size_t)(m0 + lr) * D_IN + ls;
    const bf16* Bb = Bm + (size_t)(n0 + lr) * D_IN + ls;

    const unsigned la0 = lds_off32(&As[0][lr][ls]);
    const unsigned la1 = lds_off32(&As[0][lr + 64][ls]);
    const unsigned lb0 = lds_off32(&Bs[0][lr][ls]);
    const unsigned lb1 = lds_off32(&Bs[0][lr + 64][ls]);
    const unsigned BUFSZ = 128 * 32 * 2;    // 8192 B per buffer

    auto stage = [&](int kt, int buf) {
        const unsigned bo = buf ? BUFSZ : 0u;
        const bf16* pa = Ab + kt * 32;
        const bf16* pb = Bb + kt * 32;
        async_copy16(la0 + bo, pa);
        async_copy16(la1 + bo, pa + (size_t)64 * D_IN);
        async_copy16(lb0 + bo, pb);
        async_copy16(lb1 + bo, pb + (size_t)64 * D_IN);
    };

    v8f acc[4][2] = {};
    const int ml = lane & 15;       // row/col within 16
    const int hh = lane >> 4;       // half select

    stage(0, 0);
    wait_async0();
    __syncthreads();

    const int KT = D_IN / 32;
    for (int kt = 0; kt < KT; ++kt) {
        const int buf = kt & 1;
        if (kt + 1 < KT) stage(kt + 1, buf ^ 1);   // DMA next tile during math
        if (kt + 2 < KT) {                         // warm L2 two tiles ahead
            __builtin_prefetch((const void*)(Ab + (kt + 2) * 32), 0, 1);
            __builtin_prefetch((const void*)(Bb + (kt + 2) * 32), 0, 1);
        }

        union { v16bf v; uint4 q[2]; } fa[4], fb[2];
        // A frag (16x32 bf16): lane ml holds row M; K = g*16 + hh*8 + w
#pragma unroll
        for (int mi = 0; mi < 4; ++mi) {
            const bf16* p = &As[buf][wm * 64 + mi * 16 + ml][hh * 8];
            fa[mi].q[0] = *(const uint4*)p;         // K = hh*8 + 0..7
            fa[mi].q[1] = *(const uint4*)(p + 16);  // K = 16 + hh*8 + 0..7
        }
        // B frag (32x16 bf16): lane ml holds col N; K = g*16 + hh*8 + w
#pragma unroll
        for (int ni = 0; ni < 2; ++ni) {
            const bf16* p = &Bs[buf][wn * 32 + ni * 16 + ml][hh * 8];
            fb[ni].q[0] = *(const uint4*)p;         // K = hh*8 + 0..7
            fb[ni].q[1] = *(const uint4*)(p + 16);  // K = 16 + hh*8 + 0..7
        }
#pragma unroll
        for (int mi = 0; mi < 4; ++mi)
#pragma unroll
            for (int ni = 0; ni < 2; ++ni)
                acc[mi][ni] = __builtin_amdgcn_wmma_f32_16x16x32_bf16(
                    false, fa[mi].v, false, fb[ni].v,
                    (short)0, acc[mi][ni], false, false);

        if (kt + 1 < KT) wait_async0();  // own DMA done; barrier covers others
        __syncthreads();
    }

    // epilogue: C layout — lane ml = N, VGPR r -> M = r + 8*hh ; +b_enc, ReLU
#pragma unroll
    for (int mi = 0; mi < 4; ++mi) {
#pragma unroll
        for (int ni = 0; ni < 2; ++ni) {
            const int gn  = n0 + wn * 32 + ni * 16 + ml;
            const float be = benc[gn];
#pragma unroll
            for (int r = 0; r < 8; ++r) {
                const int gm = m0 + wm * 64 + mi * 16 + r + 8 * hh;
                float v = acc[mi][ni][r] + be;
                pre[(size_t)gm * D_SAE + gn] = v > 0.f ? v : 0.f;
            }
        }
    }
}

// ---------------------------------------------------------------------------
// Kernel 4: per-row top-100 via 3-level radix select on float bits
// (all values >= 0 after ReLU, so uint bit order == float order)
// ---------------------------------------------------------------------------
__global__ __launch_bounds__(256) void k_topk(const float* __restrict__ pre,
                                              float* __restrict__ tv,
                                              int* __restrict__ ti) {
    const int row = blockIdx.x;
    const int t   = threadIdx.x;
    const float* p = pre + (size_t)row * D_SAE;

    __shared__ unsigned hist[256];
    __shared__ unsigned sSel, sCnt, sPosGT, sPosEQ;

    unsigned prefix = 0;   // matched high bytes so far
    unsigned cntGT  = 0;   // elements strictly greater than prefix

    for (int pass = 0; pass < 3; ++pass) {
        hist[t] = 0;
        __syncthreads();
        const int sh = 24 - pass * 8;
        for (int i = t; i < D_SAE; i += 256) {
            const unsigned b = __float_as_uint(p[i]);
            if (pass == 0 || (b >> (sh + 8)) == prefix)
                atomicAdd(&hist[(b >> sh) & 0xFF], 1u);
        }
        __syncthreads();
        if (t == 0) {
            unsigned c = cntGT;
            int bb;
            for (bb = 255; bb > 0; --bb) {
                if (c + hist[bb] >= KTOP) break;
                c += hist[bb];
            }
            sSel = (unsigned)bb;
            sCnt = c;
        }
        __syncthreads();
        prefix = (prefix << 8) | sSel;
        cntGT  = sCnt;
    }

    if (t == 0) { sPosGT = 0; sPosEQ = 0; }
    __syncthreads();

    float* ov = tv + (size_t)row * KTOP;
    int*   oi = ti + (size_t)row * KTOP;
    const unsigned cgt = cntGT;        // # strictly greater than 24-bit prefix
    for (int i = t; i < D_SAE; i += 256) {
        const float v   = p[i];
        const unsigned b = __float_as_uint(v) >> 8;
        if (b > prefix) {
            const unsigned pos = atomicAdd(&sPosGT, 1u);
            ov[pos] = v; oi[pos] = i;
        } else if (b == prefix) {
            const unsigned pos = atomicAdd(&sPosEQ, 1u);
            if (pos < KTOP - cgt) { ov[cgt + pos] = v; oi[cgt + pos] = i; }
        }
    }
}

// ---------------------------------------------------------------------------
// Kernel 5: sparse decode  x_hat[row] = b_dec + sum_j val_j * W_dec[idx_j, :]
// one block per row; 256 threads x 8 cols (float4 pairs); W_dec is L2-resident
// ---------------------------------------------------------------------------
__global__ __launch_bounds__(256) void k_decode(const float* __restrict__ tv,
                                                const int* __restrict__ ti,
                                                const float* __restrict__ Wdec,
                                                const float* __restrict__ bdec,
                                                float* __restrict__ out) {
    const int row = blockIdx.x;
    const int t   = threadIdx.x;
    __shared__ float sv[KTOP];
    __shared__ int   si[KTOP];
    if (t < KTOP) {
        sv[t] = tv[(size_t)row * KTOP + t];
        si[t] = ti[(size_t)row * KTOP + t];
    }
    __syncthreads();

    const int c0 = t * 8;
    float4 a0 = *(const float4*)(bdec + c0);
    float4 a1 = *(const float4*)(bdec + c0 + 4);
    for (int j = 0; j < KTOP; ++j) {
        const float v = sv[j];
        const float* wr = Wdec + (size_t)si[j] * D_IN + c0;
        const float4 w0 = *(const float4*)wr;
        const float4 w1 = *(const float4*)(wr + 4);
        a0.x = fmaf(v, w0.x, a0.x); a0.y = fmaf(v, w0.y, a0.y);
        a0.z = fmaf(v, w0.z, a0.z); a0.w = fmaf(v, w0.w, a0.w);
        a1.x = fmaf(v, w1.x, a1.x); a1.y = fmaf(v, w1.y, a1.y);
        a1.z = fmaf(v, w1.z, a1.z); a1.w = fmaf(v, w1.w, a1.w);
    }
    *(float4*)(out + (size_t)row * D_IN + c0)     = a0;
    *(float4*)(out + (size_t)row * D_IN + c0 + 4) = a1;
}

// ---------------------------------------------------------------------------
extern "C" void kernel_launch(void* const* d_in, const int* in_sizes, int n_in,
                              void* d_out, int out_size, void* d_ws, size_t ws_size,
                              hipStream_t stream) {
    const float* x     = (const float*)d_in[0];
    const float* W_enc = (const float*)d_in[1];
    const float* b_enc = (const float*)d_in[2];
    const float* W_dec = (const float*)d_in[3];
    const float* b_dec = (const float*)d_in[4];
    float* out = (float*)d_out;

    char* ws = (char*)d_ws;
    // workspace layout
    const size_t offXb  = 0;                                     // 32 MB
    const size_t offWT  = offXb + (size_t)NBATCH * D_IN * 2;     // 64 MB
    const size_t offPre = offWT + (size_t)D_SAE * D_IN * 2;      // 512 MB
    const size_t offTV  = offPre + (size_t)NBATCH * D_SAE * 4;   // 3.2 MB
    const size_t offTI  = offTV + (size_t)NBATCH * KTOP * 4;     // 3.2 MB

    bf16*  xb  = (bf16*)(ws + offXb);
    bf16*  wT  = (bf16*)(ws + offWT);
    float* pre = (float*)(ws + offPre);
    float* tv  = (float*)(ws + offTV);
    int*   ti  = (int*)(ws + offTI);

    k_convert_x <<<(NBATCH * D_IN) / (256 * 4), 256, 0, stream>>>(x, b_dec, xb);
    k_convert_wT<<<dim3(D_SAE / 64, D_IN / 64), 256, 0, stream>>>(W_enc, wT);
    k_encode_gemm<<<dim3(D_SAE / 128, NBATCH / 128), 256, 0, stream>>>(xb, wT, b_enc, pre);
    k_topk  <<<NBATCH, 256, 0, stream>>>(pre, tv, ti);
    k_decode<<<NBATCH, 256, 0, stream>>>(tv, ti, W_dec, b_dec, out);
}
